// HybridMoE_68899865362467
// MI455X (gfx1250) — compile-verified, matrix operand order checked by reading, not run on hardware
//
#include <hip/hip_runtime.h>
#include <hip/hip_bf16.h>

typedef __attribute__((ext_vector_type(16))) _Float16 v16h;
typedef __attribute__((ext_vector_type(8)))  _Float16 v8h;
typedef __attribute__((ext_vector_type(8)))  float    v8f;

#define D_DIM 1024
#define H_DIM 512
#define E_NUM 8
#define M_TOK 2048
#define T_DIM 2048
#define BT    4096

// ---------------------------------------------------------------------------
// f32 -> f16 conversion
// ---------------------------------------------------------------------------
__global__ __launch_bounds__(256) void f32_to_f16_kernel(const float* __restrict__ src,
                                                         _Float16* __restrict__ dst, int n) {
  int i = blockIdx.x * 256 + threadIdx.x;
  if (i < n) dst[i] = (_Float16)src[i];
}

__global__ void zero_counts_kernel(int* counts) {
  if (threadIdx.x < E_NUM) counts[threadIdx.x] = 0;
}

// ---------------------------------------------------------------------------
// WMMA helpers (16x16x32 f16, f32 accumulate)
// lane L: row = L%16 ; K chunks (L/16)*8..+7 and 16+(L/16)*8..+7
// ---------------------------------------------------------------------------
__device__ __forceinline__ v16h load_frag(const _Float16* p) {
  union { v16h v; v8h h[2]; } u;
  u.h[0] = *(const v8h*)p;
  u.h[1] = *(const v8h*)(p + 16);
  return u.v;
}

__device__ __forceinline__ v8f wmma16(v16h a, v16h b, v8f c) {
  return __builtin_amdgcn_wmma_f32_16x16x32_f16(false, a, false, b, (short)0, c, false, false);
}

// ---------------------------------------------------------------------------
// Fused up-projection + SiLU-gate + LayerNorm over a 32-token tile.
// As: LDS 32 x 1024 f16.  W1,W2: H x D row-major f16.  Gs: LDS 32 x 512 f32.
// Zout: 32 x 512 f16 (LDS or global), row stride zstride.
// 256 threads = 8 waves; wave w owns columns [64w, 64w+64).
// Each B fragment feeds 2 WMMAs (both M sub-tiles); B loads are double-
// buffered, and A fragments are double-buffered across K-steps so neither
// LDS nor L2 latency sits on the WMMA critical path.
// ---------------------------------------------------------------------------
__device__ __forceinline__ void ffn_up_ln32(
    const _Float16* __restrict__ As,
    const _Float16* __restrict__ W1,
    const _Float16* __restrict__ W2,
    const float* __restrict__ b1v,
    const float* __restrict__ b2v,
    const float* __restrict__ lng,
    const float* __restrict__ lnb,
    float* __restrict__ Gs,
    _Float16* Zout, int zstride) {
  const int tid  = threadIdx.x;
  const int wave = tid >> 5;
  const int lane = tid & 31;
  const int nbase = wave * 64;
  const int r  = lane & 15;
  const int kg = (lane >> 4) << 3;

  v8f c1[2][4] = {};
  v8f c2[2][4] = {};

  v16h a0 = load_frag(As + r * D_DIM + kg);
  v16h a1 = load_frag(As + (16 + r) * D_DIM + kg);

  for (int k0 = 0; k0 < D_DIM; k0 += 32) {
    const int kn = (k0 + 32) & (D_DIM - 1);   // wraps to 0 on last iter (discarded)
    const _Float16* w1p = W1 + (size_t)(nbase + r) * D_DIM + k0 + kg;
    const _Float16* w2p = W2 + (size_t)(nbase + r) * D_DIM + k0 + kg;
    v16h bcur = load_frag(w1p);
    v16h a0n = load_frag(As + r * D_DIM + kn + kg);        // next-K A, in flight
    v16h a1n = load_frag(As + (16 + r) * D_DIM + kn + kg); // next-K A, in flight
#pragma unroll
    for (int t = 0; t < 4; ++t) {
      v16h bw2 = load_frag(w2p + (size_t)t * 16 * D_DIM);  // in flight
      c1[0][t] = wmma16(a0, bcur, c1[0][t]);
      c1[1][t] = wmma16(a1, bcur, c1[1][t]);
      v16h bn = bcur;
      if (t < 3) bn = load_frag(w1p + (size_t)(t + 1) * 16 * D_DIM);  // in flight
      c2[0][t] = wmma16(a0, bw2, c2[0][t]);
      c2[1][t] = wmma16(a1, bw2, c2[1][t]);
      bcur = bn;
    }
    a0 = a0n;
    a1 = a1n;
  }

  const int mofs = (lane >> 4) << 3;   // row offset 0 or 8
  const int ncol = lane & 15;
#pragma unroll
  for (int mt = 0; mt < 2; ++mt) {
#pragma unroll
    for (int t = 0; t < 4; ++t) {
      const int n = nbase + t * 16 + ncol;
      const float bb1 = b1v[n];
      const float bb2 = b2v[n];
#pragma unroll
      for (int j = 0; j < 8; ++j) {
        float v1 = c1[mt][t][j] + bb1;
        float v2 = c2[mt][t][j] + bb2;
        float sig = 1.0f / (1.0f + __expf(-v1));
        Gs[(mt * 16 + mofs + j) * H_DIM + n] = v1 * sig * v2;
      }
    }
  }
  __syncthreads();   // Gs complete; all As reads done (Zout may alias As)

  // LayerNorm over H=512: wave w handles rows 4w..4w+3; 16 elems/lane.
#pragma unroll
  for (int rr = 0; rr < 4; ++rr) {
    const int rowi = wave * 4 + rr;
    float vals[16];
    float s = 0.0f;
#pragma unroll
    for (int i = 0; i < 16; ++i) {
      float v = Gs[rowi * H_DIM + lane + i * 32];
      vals[i] = v;
      s += v;
    }
#pragma unroll
    for (int off = 16; off >= 1; off >>= 1) s += __shfl_xor(s, off, 32);
    const float mu = s * (1.0f / (float)H_DIM);
    float var = 0.0f;
#pragma unroll
    for (int i = 0; i < 16; ++i) { float d = vals[i] - mu; var += d * d; }
#pragma unroll
    for (int off = 16; off >= 1; off >>= 1) var += __shfl_xor(var, off, 32);
    const float rstd = rsqrtf(var * (1.0f / (float)H_DIM) + 1e-5f);
#pragma unroll
    for (int i = 0; i < 16; ++i) {
      const int cc = lane + i * 32;
      float y = (vals[i] - mu) * rstd * lng[cc] + lnb[cc];
      Zout[(size_t)rowi * zstride + cc] = (_Float16)y;
    }
  }
}

// ---------------------------------------------------------------------------
// Down projection over a 32-token tile: Z (32 x 512 f16) @ W3^T, W3: D x H.
// Wave w owns columns [128w, 128w+128) -> 8 N-tiles x 2 M-tiles accumulators.
// A fragments double-buffered across K-steps; B double-buffered within.
// ---------------------------------------------------------------------------
__device__ __forceinline__ void ffn_down32(const _Float16* Zs, int zstride,
                                           const _Float16* __restrict__ W3,
                                           v8f (*cacc)[8]) {
  const int tid  = threadIdx.x;
  const int wave = tid >> 5;
  const int lane = tid & 31;
  const int nbase = wave * 128;
  const int r  = lane & 15;
  const int kg = (lane >> 4) << 3;

  v16h a0 = load_frag(Zs + (size_t)r * zstride + kg);
  v16h a1 = load_frag(Zs + (size_t)(16 + r) * zstride + kg);

  for (int k0 = 0; k0 < H_DIM; k0 += 32) {
    const int kn = (k0 + 32) & (H_DIM - 1);   // wraps to 0 on last iter (discarded)
    const _Float16* wp = W3 + (size_t)(nbase + r) * H_DIM + k0 + kg;
    v16h bcur = load_frag(wp);
    v16h a0n = load_frag(Zs + (size_t)r * zstride + kn + kg);        // in flight
    v16h a1n = load_frag(Zs + (size_t)(16 + r) * zstride + kn + kg); // in flight
#pragma unroll
    for (int t = 0; t < 8; ++t) {
      v16h bn = bcur;
      if (t < 7) bn = load_frag(wp + (size_t)(t + 1) * 16 * H_DIM);  // in flight
      cacc[0][t] = wmma16(a0, bcur, cacc[0][t]);
      cacc[1][t] = wmma16(a1, bcur, cacc[1][t]);
      bcur = bn;
    }
    a0 = a0n;
    a1 = a1n;
  }
}

// ---------------------------------------------------------------------------
// Shared expert: up projection + LN -> sh (BT x H, f16).  32 tokens / block.
// dynamic LDS: [0,64KB) A-tile f16, [64KB,128KB) G f32
// ---------------------------------------------------------------------------
__global__ __launch_bounds__(256) void shared_ffn_up_kernel(
    const _Float16* __restrict__ xh,
    const _Float16* __restrict__ w1h, const _Float16* __restrict__ w2h,
    const float* __restrict__ b1, const float* __restrict__ b2,
    const float* __restrict__ lng, const float* __restrict__ lnb,
    _Float16* __restrict__ sh) {
  extern __shared__ __align__(16) unsigned char dynsmem[];
  _Float16* As = (_Float16*)dynsmem;           // 64 KB
  float*    Gs = (float*)(dynsmem + 65536);    // 64 KB
  const int row0 = blockIdx.x * 32;
  const int4* src = (const int4*)(xh + (size_t)row0 * D_DIM);
  int4* dst = (int4*)As;
  for (int i = threadIdx.x; i < (32 * D_DIM) / 8; i += 256) dst[i] = src[i];
  __syncthreads();
  ffn_up_ln32(As, w1h, w2h, b1, b2, lng, lnb, Gs, sh + (size_t)row0 * H_DIM, H_DIM);
}

// ---------------------------------------------------------------------------
// Shared expert: down projection -> out (BT x D, f32) == shared_out
// ---------------------------------------------------------------------------
__global__ __launch_bounds__(256) void shared_ffn_down_kernel(
    const _Float16* __restrict__ sh, const _Float16* __restrict__ w3h,
    const float* __restrict__ b3, float* __restrict__ out) {
  extern __shared__ __align__(16) unsigned char dynsmem[];
  _Float16* Zs = (_Float16*)dynsmem;           // 32 KB
  const int row0 = blockIdx.x * 32;
  const int4* src = (const int4*)(sh + (size_t)row0 * H_DIM);
  int4* dst = (int4*)Zs;
  for (int i = threadIdx.x; i < (32 * H_DIM) / 8; i += 256) dst[i] = src[i];
  __syncthreads();
  v8f c[2][8] = {};
  ffn_down32(Zs, H_DIM, w3h, c);
  const int wave = threadIdx.x >> 5, lane = threadIdx.x & 31;
  const int mofs = (lane >> 4) << 3, ncol = lane & 15;
#pragma unroll
  for (int mt = 0; mt < 2; ++mt) {
#pragma unroll
    for (int t = 0; t < 8; ++t) {
      const int n = wave * 128 + t * 16 + ncol;
      const float bb = b3[n];
#pragma unroll
      for (int j = 0; j < 8; ++j)
        out[(size_t)(row0 + mt * 16 + mofs + j) * D_DIM + n] = c[mt][t][j] + bb;
    }
  }
}

// ---------------------------------------------------------------------------
// Gating + routing: one block per MoE token.
// ---------------------------------------------------------------------------
__global__ __launch_bounds__(256) void gate_route_kernel(
    const float* __restrict__ x, const int* __restrict__ index_b,
    const int* __restrict__ index_t,
    const float* __restrict__ sg_w, const float* __restrict__ sg_b,
    const float* __restrict__ eg_w, const float* __restrict__ eg_b,
    const float* __restrict__ exp_bias,
    int* __restrict__ rowid, float* __restrict__ shw, float* __restrict__ exw,
    int* __restrict__ counts, int* __restrict__ lists) {
  const int m = blockIdx.x;
  const int tid = threadIdx.x;
  const int row = index_b[m] * T_DIM + index_t[m];
  const float* xr = x + (size_t)row * D_DIM;
  float acc[9];
#pragma unroll
  for (int i = 0; i < 9; ++i) acc[i] = 0.0f;
  for (int k = tid; k < D_DIM; k += 256) {
    const float xv = xr[k];
    acc[0] += xv * sg_w[k];
#pragma unroll
    for (int e = 0; e < E_NUM; ++e) acc[1 + e] += xv * eg_w[e * D_DIM + k];
  }
  __shared__ float red[9][8];
  const int wave = tid >> 5, lane = tid & 31;
#pragma unroll
  for (int i = 0; i < 9; ++i) {
    float v = acc[i];
#pragma unroll
    for (int off = 16; off >= 1; off >>= 1) v += __shfl_xor(v, off, 32);
    if (lane == 0) red[i][wave] = v;
  }
  __syncthreads();
  if (tid == 0) {
    float tot[9];
#pragma unroll
    for (int i = 0; i < 9; ++i) {
      float s = 0.0f;
      for (int w = 0; w < 8; ++w) s += red[i][w];
      tot[i] = s;
    }
    const float ss = 1.0f / (1.0f + __expf(-(tot[0] + sg_b[0])));
    int best = 0; float bestkey = -1e30f; float bestscore = 0.0f;
#pragma unroll
    for (int e = 0; e < E_NUM; ++e) {            // first-occurrence argmax
      float sc = 1.0f / (1.0f + __expf(-(tot[1 + e] + eg_b[e])));
      float key = sc + exp_bias[e];
      if (key > bestkey) { bestkey = key; best = e; bestscore = sc; }
    }
    const float mx = fmaxf(ss, bestscore);
    const float e1 = __expf(ss - mx);
    const float e2 = __expf(bestscore - mx);
    const float inv = 1.0f / (e1 + e2);
    rowid[m] = row;
    shw[m] = e1 * inv;
    exw[m] = e2 * inv;
    int slot = atomicAdd(&counts[best], 1);
    lists[(size_t)best * M_TOK + slot] = m;
  }
}

// ---------------------------------------------------------------------------
// Routed-expert FFN: grid (tiles, experts); each block = 32 tokens of expert e.
// out[row] = shared_out[row]*w_s + (expert_out + b3)*w_e
// dynamic LDS: [0,64KB) A-tile f16 (Z overlays it), [64KB,128KB) G f32
// ---------------------------------------------------------------------------
__global__ __launch_bounds__(256) void expert_ffn_kernel(
    const _Float16* __restrict__ xh,
    const _Float16* __restrict__ ew1h, const _Float16* __restrict__ ew2h,
    const _Float16* __restrict__ ew3h,
    const float* __restrict__ eb1, const float* __restrict__ eb2,
    const float* __restrict__ eb3,
    const float* __restrict__ lng, const float* __restrict__ lnb,
    const int* __restrict__ counts, const int* __restrict__ lists,
    const int* __restrict__ rowid,
    const float* __restrict__ shw, const float* __restrict__ exw,
    float* __restrict__ out) {
  const int e   = blockIdx.y;
  const int cnt = counts[e];
  const int t0  = blockIdx.x * 32;
  if (t0 >= cnt) return;

  extern __shared__ __align__(16) unsigned char dynsmem[];
  _Float16* As = (_Float16*)dynsmem;            // 64 KB (dead after GEMM1)
  float*    Gs = (float*)(dynsmem + 65536);     // 64 KB
  _Float16* Zs = (_Float16*)dynsmem;            // 32 KB, overlays As
  __shared__ int   rows_s[32];
  __shared__ int   valid_s[32];
  __shared__ float sw_s[32], ew_s[32];

  const int tid = threadIdx.x;
  if (tid < 32) {
    const int idx = t0 + tid;
    const int ok  = idx < cnt;
    const int m   = lists[(size_t)e * M_TOK + (ok ? idx : t0)];
    rows_s[tid]  = rowid[m];
    valid_s[tid] = ok;
    sw_s[tid]    = shw[m];
    ew_s[tid]    = exw[m];
  }
  __syncthreads();

  // Gather 32 token rows (row-major 32x1024 f16) into LDS.
  for (int v = tid; v < (32 * D_DIM) / 8; v += 256) {
    const int r = v >> 7, c = v & 127;
    ((int4*)As)[v] = ((const int4*)(xh + (size_t)rows_s[r] * D_DIM))[c];
  }
  __syncthreads();

  ffn_up_ln32(As, ew1h + (size_t)e * H_DIM * D_DIM, ew2h + (size_t)e * H_DIM * D_DIM,
              eb1 + e * H_DIM, eb2 + e * H_DIM, lng, lnb, Gs, Zs, H_DIM);
  __syncthreads();

  v8f c[2][8] = {};
  ffn_down32(Zs, H_DIM, ew3h + (size_t)e * D_DIM * H_DIM, c);

  const int wave = tid >> 5, lane = tid & 31;
  const int mofs = (lane >> 4) << 3, ncol = lane & 15;
#pragma unroll
  for (int mt = 0; mt < 2; ++mt) {
#pragma unroll
    for (int t = 0; t < 8; ++t) {
      const int n = wave * 128 + t * 16 + ncol;
      const float bb = eb3[e * D_DIM + n];
#pragma unroll
      for (int j = 0; j < 8; ++j) {
        const int mr = mt * 16 + mofs + j;
        if (valid_s[mr]) {
          const size_t o = (size_t)rows_s[mr] * D_DIM + n;
          const float val = c[mt][t][j] + bb;
          out[o] = out[o] * sw_s[mr] + val * ew_s[mr];
        }
      }
    }
  }
}

// ---------------------------------------------------------------------------
extern "C" void kernel_launch(void* const* d_in, const int* in_sizes, int n_in,
                              void* d_out, int out_size, void* d_ws, size_t ws_size,
                              hipStream_t stream) {
  const float* x        = (const float*)d_in[0];
  const int*   index_b  = (const int*)d_in[1];
  const int*   index_t  = (const int*)d_in[2];
  const float* sw1      = (const float*)d_in[3];
  const float* sb1      = (const float*)d_in[4];
  const float* sw2      = (const float*)d_in[5];
  const float* sb2      = (const float*)d_in[6];
  const float* sw3      = (const float*)d_in[7];
  const float* sb3      = (const float*)d_in[8];
  const float* s_ln_g   = (const float*)d_in[9];
  const float* s_ln_b   = (const float*)d_in[10];
  const float* sg_w     = (const float*)d_in[11];
  const float* sg_b     = (const float*)d_in[12];
  const float* eg_w     = (const float*)d_in[13];
  const float* eg_b     = (const float*)d_in[14];
  const float* exp_bias = (const float*)d_in[15];
  const float* ew1      = (const float*)d_in[16];
  const float* eb1      = (const float*)d_in[17];
  const float* ew2      = (const float*)d_in[18];
  const float* eb2      = (const float*)d_in[19];
  const float* ew3      = (const float*)d_in[20];
  const float* eb3      = (const float*)d_in[21];
  const float* sel_ln_g = (const float*)d_in[22];
  const float* sel_ln_b = (const float*)d_in[23];
  float* out = (float*)d_out;

  unsigned char* ws = (unsigned char*)d_ws;
  size_t off = 0;
  auto take = [&](size_t bytes) -> void* {
    void* p = ws + off;
    off += (bytes + 255) & ~(size_t)255;
    return p;
  };
  _Float16* xh   = (_Float16*)take((size_t)BT * D_DIM * 2);
  _Float16* sw1h = (_Float16*)take((size_t)H_DIM * D_DIM * 2);
  _Float16* sw2h = (_Float16*)take((size_t)H_DIM * D_DIM * 2);
  _Float16* sw3h = (_Float16*)take((size_t)D_DIM * H_DIM * 2);
  _Float16* ew1h = (_Float16*)take((size_t)E_NUM * H_DIM * D_DIM * 2);
  _Float16* ew2h = (_Float16*)take((size_t)E_NUM * H_DIM * D_DIM * 2);
  _Float16* ew3h = (_Float16*)take((size_t)E_NUM * D_DIM * H_DIM * 2);
  _Float16* shb  = (_Float16*)take((size_t)BT * H_DIM * 2);
  int*   rowid  = (int*)take((size_t)M_TOK * 4);
  float* shw    = (float*)take((size_t)M_TOK * 4);
  float* exw    = (float*)take((size_t)M_TOK * 4);
  int*   counts = (int*)take((size_t)E_NUM * 4);
  int*   lists  = (int*)take((size_t)E_NUM * M_TOK * 4);

  auto cvt = [&](const float* s, _Float16* d, int n) {
    f32_to_f16_kernel<<<(n + 255) / 256, 256, 0, stream>>>(s, d, n);
  };
  cvt(x,   xh,   BT * D_DIM);
  cvt(sw1, sw1h, H_DIM * D_DIM);
  cvt(sw2, sw2h, H_DIM * D_DIM);
  cvt(sw3, sw3h, D_DIM * H_DIM);
  cvt(ew1, ew1h, E_NUM * H_DIM * D_DIM);
  cvt(ew2, ew2h, E_NUM * H_DIM * D_DIM);
  cvt(ew3, ew3h, E_NUM * D_DIM * H_DIM);

  zero_counts_kernel<<<1, 32, 0, stream>>>(counts);

  shared_ffn_up_kernel<<<BT / 32, 256, 131072, stream>>>(xh, sw1h, sw2h, sb1, sb2,
                                                         s_ln_g, s_ln_b, shb);
  shared_ffn_down_kernel<<<BT / 32, 256, 32768, stream>>>(shb, sw3h, sb3, out);

  gate_route_kernel<<<M_TOK, 256, 0, stream>>>(x, index_b, index_t, sg_w, sg_b,
                                               eg_w, eg_b, exp_bias,
                                               rowid, shw, exw, counts, lists);

  expert_ffn_kernel<<<dim3(M_TOK / 32, E_NUM), 256, 131072, stream>>>(
      xh, ew1h, ew2h, ew3h, eb1, eb2, eb3, sel_ln_g, sel_ln_b,
      counts, lists, rowid, shw, exw, out);
}
